// ConformerBlock_70222715289718
// MI455X (gfx1250) — matrix operand order repaired and checked by reading, not compiled, not run on hardware
//
#include <hip/hip_runtime.h>
#include <hip/hip_bf16.h>
#include <stdint.h>

#define DEV __device__ __forceinline__

typedef __bf16 bf16_t;
typedef bf16_t v16bf __attribute__((ext_vector_type(16)));
typedef bf16_t v8bf  __attribute__((ext_vector_type(8)));
typedef float  v8f   __attribute__((ext_vector_type(8)));

// Problem dims (fixed by the reference)
constexpr int Bb    = 4;
constexpr int Ss    = 1024;
constexpr int Hh    = 512;
constexpr int NHh   = 8;
constexpr int DHh   = 64;
constexpr int Ff    = 2048;
constexpr int Kc    = 31;
constexpr int Mrows = Bb * Ss;      // 4096
constexpr int Rpos  = 2 * Ss - 1;   // 2047
constexpr int RposPad = 2048;

DEV v8f vzero() {
  v8f z;
#pragma unroll
  for (int i = 0; i < 8; i++) z[i] = 0.f;
  return z;
}

DEV v8f wmma_bf16(v16bf a, v16bf b, v8f c) {
  return __builtin_amdgcn_wmma_f32_16x16x32_bf16(
      /*neg_a=*/false, a, /*neg_b=*/false, b,
      /*c_mod=*/(short)0, c, /*reuse_a=*/false, /*reuse_b=*/false);
}

union AB16 { v16bf v; v8bf h[2]; };

// A-fragment (16x32, bf16): lane = half*16+lc holds row lc; elements 0..7 = K
// c0..c0+7, elements 8..15 = K 16+c0..16+c0+7 where c0 = 8*half. rowptr must
// point at (row, kchunk_base) and be 16B aligned.
DEV v16bf ld_afrag(const bf16_t* rowptr, int half) {
  AB16 u;
  u.h[0] = *(const v8bf*)(rowptr + 8 * half);
  u.h[1] = *(const v8bf*)(rowptr + 16 + 8 * half);
  return u.v;
}

// B-fragment (32x16, bf16): lane holds column n = lc; elements 0..7 = K
// c0..c0+7, elements 8..15 = K 16+c0..16+c0+7 where c0 = 8*half.
// colptr = &B[n][kchunk_base] (n-major storage).
DEV v16bf ld_bfrag(const bf16_t* colptr, int half) {
  AB16 u;
  u.h[0] = *(const v8bf*)(colptr + 8 * half);
  u.h[1] = *(const v8bf*)(colptr + 16 + 8 * half);
  return u.v;
}

DEV float sigmoidf_(float x) { return 1.f / (1.f + __expf(-x)); }

// ---------------------------------------------------------------------------
// LayerNorm over H=512 per row; one wave32 per row (16 elems/lane).
// Optionally writes bf16 output, fp32 output, and a raw fp32 copy of input.
// ---------------------------------------------------------------------------
__global__ __launch_bounds__(256) void ln_k(const float* __restrict__ in,
                                            const float* __restrict__ g,
                                            const float* __restrict__ bta,
                                            bf16_t* obf, float* of32,
                                            float* xcopy, int rows) {
  int row  = blockIdx.x * 8 + (threadIdx.x >> 5);
  int lane = threadIdx.x & 31;
  if (row >= rows) return;
  const float* p = in + (size_t)row * Hh;
  float vals[16];
  float s = 0.f;
#pragma unroll
  for (int i = 0; i < 16; i++) { vals[i] = p[lane + 32 * i]; s += vals[i]; }
#pragma unroll
  for (int o = 16; o >= 1; o >>= 1) s += __shfl_xor(s, o, 32);
  float mean = s * (1.f / Hh);
  float vs = 0.f;
#pragma unroll
  for (int i = 0; i < 16; i++) { float d = vals[i] - mean; vs += d * d; }
#pragma unroll
  for (int o = 16; o >= 1; o >>= 1) vs += __shfl_xor(vs, o, 32);
  float rinv = rsqrtf(vs * (1.f / Hh) + 1e-5f);
#pragma unroll
  for (int i = 0; i < 16; i++) {
    int c = lane + 32 * i;
    float y = (vals[i] - mean) * rinv * g[c] + bta[c];
    size_t idx = (size_t)row * Hh + c;
    if (obf)   obf[idx]   = (bf16_t)y;
    if (of32)  of32[idx]  = y;
    if (xcopy) xcopy[idx] = vals[i];
  }
}

// ---------------------------------------------------------------------------
// WMMA GEMM: C[M,N] = A[M,K] x B[K,N] (+bias) with fused epilogues.
//   EPI 0: out bf16 = acc + bias
//   EPI 1: out bf16 = silu(acc + bias)
//   EPI 2: xres     += alpha * (acc + bias)     (fp32 residual accumulate)
//   EPI 3: obf = bf16(acc+bias+pu[n]); obf2 = bf16(acc+bias+pv[n])  (q split)
// TRANSB: B stored [N][K] row-major (for pw1/pw2: h @ W^T)
// A32:    A stored fp32 (converted to bf16 while staging to LDS)
// Tile: 64x128, BK=64 (two WMMA k-chunks per stage); 256 threads = 8 waves
// in 2x4; 32x32 per wave; 8 WMMAs per wave per stage.
// B global traffic is coalesced along N (8 contiguous floats per thread);
// the [K,N] -> [n][k] transpose happens on the LDS-write side.
// ---------------------------------------------------------------------------
template <int EPI, bool TRANSB, bool A32>
__global__ __launch_bounds__(256) void gemm_k(
    const void* __restrict__ Av, const float* __restrict__ Bw,
    const float* __restrict__ bias, bf16_t* __restrict__ obf,
    float* __restrict__ ores, const float* __restrict__ pu,
    const float* __restrict__ pv, bf16_t* __restrict__ obf2,
    int Md, int Nd, int Kd, float alpha) {
  __shared__ bf16_t As[64][64];     // [m][k]
  __shared__ bf16_t Bs[128][64];    // [n][k]

  int tid  = threadIdx.x;
  int lane = tid & 31, wave = tid >> 5;
  int half = lane >> 4, lc = lane & 15;
  int m0 = blockIdx.y * 64;
  int n0 = blockIdx.x * 128;
  int wm = wave >> 2;      // 0..1
  int wn = wave & 3;       // 0..3

  v8f acc[2][2];
#pragma unroll
  for (int i = 0; i < 2; i++)
#pragma unroll
    for (int j = 0; j < 2; j++) acc[i][j] = vzero();

  // A staging coords: 64 rows x 64 k = 4096 elems, 16 per thread
  int ar = tid >> 2;            // 0..63
  int ak = (tid & 3) * 16;      // 0,16,32,48
  int grow = m0 + ar;
  // B staging coords (!TRANSB): coalesced 8-float reads along N
  int bkr = tid >> 4;           // 0..15  (k row, 4 passes of +16)
  int bn8 = (tid & 15) * 8;     // 0..120 (n chunk)
  // B staging coords (TRANSB): contiguous k reads
  int btn = tid >> 1;           // 0..127
  int btk = (tid & 1) * 32;     // 0,32

  for (int kt = 0; kt < Kd; kt += 64) {
    // ---- stage A (convert to bf16 if needed) ----
    if (A32) {
      const float* Af = (const float*)Av;
      if (grow < Md) {
        const float* src = Af + (size_t)grow * Kd + kt + ak;
#pragma unroll
        for (int i = 0; i < 16; i++) As[ar][ak + i] = (bf16_t)src[i];
      } else {
#pragma unroll
        for (int i = 0; i < 16; i++) As[ar][ak + i] = (bf16_t)0.f;
      }
    } else {
      const bf16_t* Ab = (const bf16_t*)Av;
      if (grow < Md) {
        const bf16_t* src = Ab + (size_t)grow * Kd + kt + ak;
        *(v8bf*)&As[ar][ak]     = *(const v8bf*)(src);
        *(v8bf*)&As[ar][ak + 8] = *(const v8bf*)(src + 8);
      } else {
#pragma unroll
        for (int i = 0; i < 16; i++) As[ar][ak + i] = (bf16_t)0.f;
      }
    }
    // ---- stage B into n-major [n][k] LDS ----
    if (TRANSB) {
      // B stored [N][K]: contiguous along k, straight copy
      const float* src = Bw + (size_t)(n0 + btn) * Kd + kt + btk;
#pragma unroll
      for (int i = 0; i < 32; i++) Bs[btn][btk + i] = (bf16_t)src[i];
    } else {
      // B stored [K,N]: coalesced global reads along N, transpose into LDS
#pragma unroll
      for (int p = 0; p < 4; p++) {
        int k = bkr + 16 * p;
        const float* src = Bw + (size_t)(kt + k) * Nd + n0 + bn8;
#pragma unroll
        for (int i = 0; i < 8; i++) Bs[bn8 + i][k] = (bf16_t)src[i];
      }
    }
    __syncthreads();

#pragma unroll
    for (int kc = 0; kc < 2; kc++) {
      v16bf a0 = ld_afrag(&As[wm * 32 + lc][kc * 32], half);
      v16bf a1 = ld_afrag(&As[wm * 32 + 16 + lc][kc * 32], half);
      v16bf b0 = ld_bfrag(&Bs[wn * 32 + lc][kc * 32], half);
      v16bf b1 = ld_bfrag(&Bs[wn * 32 + 16 + lc][kc * 32], half);
      acc[0][0] = wmma_bf16(a0, b0, acc[0][0]);
      acc[0][1] = wmma_bf16(a0, b1, acc[0][1]);
      acc[1][0] = wmma_bf16(a1, b0, acc[1][0]);
      acc[1][1] = wmma_bf16(a1, b1, acc[1][1]);
    }
    __syncthreads();
  }

  // ---- epilogue ----
#pragma unroll
  for (int rs = 0; rs < 2; rs++) {
#pragma unroll
    for (int cs = 0; cs < 2; cs++) {
      int gc = n0 + wn * 32 + cs * 16 + lc;
      float bv = bias ? bias[gc] : 0.f;
#pragma unroll
      for (int v = 0; v < 8; v++) {
        int gr = m0 + wm * 32 + rs * 16 + v + 8 * half;
        if (gr >= Md) continue;
        float val = acc[rs][cs][v] + bv;
        size_t idx = (size_t)gr * Nd + gc;
        if (EPI == 0) {
          obf[idx] = (bf16_t)val;
        } else if (EPI == 1) {
          obf[idx] = (bf16_t)(val * sigmoidf_(val));
        } else if (EPI == 2) {
          ores[idx] += alpha * val;
        } else if (EPI == 3) {
          obf[idx]  = (bf16_t)(val + pu[gc]);
          obf2[idx] = (bf16_t)(val + pv[gc]);
        }
      }
    }
  }
}

// ---------------------------------------------------------------------------
// Fused relative-position flash attention.
// Grid: (S/64, B*NH); block 128 (4 waves); wave w owns 16 query rows.
// scores = ((q+pu)K^T + rel_shift((q+pv)P^T)) / sqrt(DH); online softmax;
// O = P V. bd handled per 16x16 tile: two WMMAs vs 32 consecutive position
// rows + per-register diagonal gather via __shfl.
// ---------------------------------------------------------------------------
__global__ __launch_bounds__(128) void attn_k(
    const bf16_t* __restrict__ qu, const bf16_t* __restrict__ qv,
    const bf16_t* __restrict__ kb, const bf16_t* __restrict__ vb,
    const bf16_t* __restrict__ pb, bf16_t* __restrict__ ob) {
  __shared__ bf16_t v_lds[64][32];        // [d][t-in-chunk], n-major for B frags
  __shared__ bf16_t p_lds[4][16][32];     // per-wave P tile, row-major

  int bn = blockIdx.y;
  int b = bn / NHh, n = bn % NHh;
  int wave = threadIdx.x >> 5, lane = threadIdx.x & 31;
  int half = lane >> 4, lc = lane & 15;
  int sbase = blockIdx.x * 64 + wave * 16;

  const bf16_t* qub = qu + ((size_t)(b * Ss + sbase)) * Hh + n * DHh;
  const bf16_t* qvb = qv + ((size_t)(b * Ss + sbase)) * Hh + n * DHh;
  const bf16_t* kbase = kb + (size_t)(b * Ss) * Hh + n * DHh;
  const bf16_t* vbase = vb + (size_t)(b * Ss) * Hh + n * DHh;
  const bf16_t* pbase = pb + n * DHh;

  v16bf aqu[2], aqv[2];
#pragma unroll
  for (int kk = 0; kk < 2; kk++) {
    aqu[kk] = ld_afrag(qub + (size_t)lc * Hh + kk * 32, half);
    aqv[kk] = ld_afrag(qvb + (size_t)lc * Hh + kk * 32, half);
  }

  v8f oacc[4];
  float m8[8], l8[8];
#pragma unroll
  for (int j = 0; j < 4; j++) oacc[j] = vzero();
#pragma unroll
  for (int v = 0; v < 8; v++) { m8[v] = -1e30f; l8[v] = 0.f; }

  for (int t0 = 0; t0 < Ss; t0 += 32) {
    __syncthreads();
    {   // cooperative V tile stage, transposed to [d][t]
      int t  = threadIdx.x >> 2;
      int d0 = (threadIdx.x & 3) * 16;
      const bf16_t* src = vbase + (size_t)(t0 + t) * Hh + d0;
#pragma unroll
      for (int i = 0; i < 16; i++) v_lds[d0 + i][t] = src[i];
    }
    __syncthreads();

    v8f st[2];
    float lsum[8];
#pragma unroll
    for (int v = 0; v < 8; v++) lsum[v] = 0.f;

#pragma unroll
    for (int ht = 0; ht < 2; ht++) {
      int tt = t0 + 16 * ht;
      // ---- content scores (q+pu)·k ----
      v8f sc = vzero();
      const bf16_t* kcol = kbase + (size_t)(tt + lc) * Hh;
      sc = wmma_bf16(aqu[0], ld_bfrag(kcol, half), sc);
      sc = wmma_bf16(aqu[1], ld_bfrag(kcol + 32, half), sc);
      // ---- position scores: E[srow, c] = (q+pv)·p[rA + c], c in [0,31] ----
      int rA = tt - sbase + (Ss - 1) - 15;   // in [0, 2S-32]
      const bf16_t* p1 = pbase + (size_t)(rA + lc) * Hh;
      const bf16_t* p2 = pbase + (size_t)(rA + 16 + lc) * Hh;
      v8f e1 = vzero(), e2 = vzero();
      e1 = wmma_bf16(aqv[0], ld_bfrag(p1, half), e1);
      e1 = wmma_bf16(aqv[1], ld_bfrag(p1 + 32, half), e1);
      e2 = wmma_bf16(aqv[0], ld_bfrag(p2, half), e2);
      e2 = wmma_bf16(aqv[1], ld_bfrag(p2 + 32, half), e2);
      // diagonal gather: bd[srow, c] = E[srow, c - srow + 15]
#pragma unroll
      for (int v = 0; v < 8; v++) {
        int srow = v + 8 * half;
        int ce   = lc - srow + 15;             // 0..30
        int srcl = half * 16 + (ce & 15);
        float b1 = __shfl(e1[v], srcl, 32);
        float b2 = __shfl(e2[v], srcl, 32);
        sc[v] = (sc[v] + ((ce < 16) ? b1 : b2)) * 0.125f;  // 1/sqrt(64)
      }
      st[ht] = sc;
    }

    // ---- online softmax update over these 32 columns ----
    float mn[8], corr[8];
#pragma unroll
    for (int v = 0; v < 8; v++) {
      float t = fmaxf(st[0][v], st[1][v]);
#pragma unroll
      for (int o = 8; o >= 1; o >>= 1) t = fmaxf(t, __shfl_xor(t, o, 32));
      mn[v]   = fmaxf(m8[v], t);
      corr[v] = __expf(m8[v] - mn[v]);
      m8[v]   = mn[v];
    }
#pragma unroll
    for (int ht = 0; ht < 2; ht++) {
#pragma unroll
      for (int v = 0; v < 8; v++) {
        float pe = __expf(st[ht][v] - mn[v]);
        p_lds[wave][v + 8 * half][ht * 16 + lc] = (bf16_t)pe;
        lsum[v] += pe;
      }
    }
#pragma unroll
    for (int v = 0; v < 8; v++) {
      float t = lsum[v];
#pragma unroll
      for (int o = 8; o >= 1; o >>= 1) t += __shfl_xor(t, o, 32);
      l8[v] = l8[v] * corr[v] + t;
    }
#pragma unroll
    for (int j = 0; j < 4; j++)
#pragma unroll
      for (int v = 0; v < 8; v++) oacc[j][v] *= corr[v];

    // ---- O += P (16x32) x V (32x64) ----
    v16bf pa = ld_afrag(&p_lds[wave][lc][0], half);
#pragma unroll
    for (int j = 0; j < 4; j++) {
      v16bf bf = ld_bfrag(&v_lds[j * 16 + lc][0], half);
      oacc[j] = wmma_bf16(pa, bf, oacc[j]);
    }
  }

  // ---- normalize and write [b*S+s][n*64+d] ----
  bf16_t* obase = ob + (size_t)(b * Ss + sbase) * Hh + n * DHh;
#pragma unroll
  for (int v = 0; v < 8; v++) {
    float inv = 1.f / l8[v];
    int r = v + 8 * half;
#pragma unroll
    for (int j = 0; j < 4; j++)
      obase[(size_t)r * Hh + j * 16 + lc] = (bf16_t)(oacc[j][v] * inv);
  }
}

// ---------------------------------------------------------------------------
// GLU over channel dim: out[m,h] = a * sigmoid(b), a=in[m,h], b=in[m,H+h]
// ---------------------------------------------------------------------------
__global__ __launch_bounds__(256) void glu_k(const bf16_t* __restrict__ in,
                                             bf16_t* __restrict__ out) {
  int idx = blockIdx.x * 256 + threadIdx.x;   // over M*H
  int m = idx / Hh, h = idx % Hh;
  float a = (float)in[(size_t)m * (2 * Hh) + h];
  float g = (float)in[(size_t)m * (2 * Hh) + Hh + h];
  out[idx] = (bf16_t)(a * sigmoidf_(g));
}

// ---------------------------------------------------------------------------
// Depthwise conv (K=31, same-pad) + eval BatchNorm + SiLU
// ---------------------------------------------------------------------------
__global__ __launch_bounds__(256) void dwconv_k(const bf16_t* __restrict__ in,
                                                const float* __restrict__ dw,
                                                const float* __restrict__ bng,
                                                const float* __restrict__ bnb,
                                                bf16_t* __restrict__ out) {
  int idx = blockIdx.x * 256 + threadIdx.x;   // over M*H
  int m = idx / Hh, h = idx % Hh;
  int b = m / Ss, s = m % Ss;
  float acc = 0.f;
#pragma unroll
  for (int j = 0; j < Kc; j++) {
    int ss = s + j - (Kc - 1) / 2;
    if (ss >= 0 && ss < Ss)
      acc += (float)in[((size_t)b * Ss + ss) * Hh + h] * dw[h * Kc + j];
  }
  float y = acc * (bng[h] * rsqrtf(1.f + 1e-5f)) + bnb[h];
  out[idx] = (bf16_t)(y * sigmoidf_(y));
}

// ---------------------------------------------------------------------------
// Host orchestration
// ---------------------------------------------------------------------------
extern "C" void kernel_launch(void* const* d_in, const int* in_sizes, int n_in,
                              void* d_out, int out_size, void* d_ws,
                              size_t ws_size, hipStream_t stream) {
  (void)in_sizes; (void)n_in; (void)out_size; (void)ws_size;

  const float* hidden   = (const float*)d_in[0];
  const float* posemb   = (const float*)d_in[1];
  const float* f1_ln_g  = (const float*)d_in[2];
  const float* f1_ln_b  = (const float*)d_in[3];
  const float* f1_w1    = (const float*)d_in[4];
  const float* f1_b1    = (const float*)d_in[5];
  const float* f1_w2    = (const float*)d_in[6];
  const float* f1_b2    = (const float*)d_in[7];
  const float* at_ln_g  = (const float*)d_in[8];
  const float* at_ln_b  = (const float*)d_in[9];
  const float* wq       = (const float*)d_in[10];
  const float* bq       = (const float*)d_in[11];
  const float* wk       = (const float*)d_in[12];
  const float* bk       = (const float*)d_in[13];
  const float* wv       = (const float*)d_in[14];
  const float* bv       = (const float*)d_in[15];
  const float* wpos     = (const float*)d_in[16];
  const float* pos_u    = (const float*)d_in[17];
  const float* pos_v    = (const float*)d_in[18];
  const float* wo       = (const float*)d_in[19];
  const float* bo       = (const float*)d_in[20];
  const float* cv_ln_g  = (const float*)d_in[21];
  const float* cv_ln_b  = (const float*)d_in[22];
  const float* pw1_w    = (const float*)d_in[23];
  const float* dw_w     = (const float*)d_in[24];
  const float* bn_g     = (const float*)d_in[25];
  const float* bn_b     = (const float*)d_in[26];
  const float* pw2_w    = (const float*)d_in[27];
  const float* f2_ln_g  = (const float*)d_in[28];
  const float* f2_ln_b  = (const float*)d_in[29];
  const float* f2_w1    = (const float*)d_in[30];
  const float* f2_b1    = (const float*)d_in[31];
  const float* f2_w2    = (const float*)d_in[32];
  const float* f2_b2    = (const float*)d_in[33];
  const float* fn_ln_g  = (const float*)d_in[34];
  const float* fn_ln_b  = (const float*)d_in[35];

  char* ws = (char*)d_ws;
  size_t off = 0;
  auto take = [&](size_t bytes) -> void* {
    void* p = ws + off;
    off = (off + bytes + 255) & ~(size_t)255;
    return p;
  };
  float*  x    = (float*) take((size_t)Mrows * Hh * 4);       // residual (fp32)
  bf16_t* hln  = (bf16_t*)take((size_t)Mrows * Hh * 2);       // LN output (bf16)
  bf16_t* t0   = (bf16_t*)take((size_t)Mrows * Ff * 2);       // FFN mid / pw1 out
  bf16_t* qu   = (bf16_t*)take((size_t)Mrows * Hh * 2);
  bf16_t* qvb  = (bf16_t*)take((size_t)Mrows * Hh * 2);
  bf16_t* kbf  = (bf16_t*)take((size_t)Mrows * Hh * 2);
  bf16_t* vbf  = (bf16_t*)take((size_t)Mrows * Hh * 2);
  bf16_t* pbf  = (bf16_t*)take((size_t)RposPad * Hh * 2);
  bf16_t* ao   = (bf16_t*)take((size_t)Mrows * Hh * 2);
  bf16_t* glu  = (bf16_t*)take((size_t)Mrows * Hh * 2);
  bf16_t* cvo  = (bf16_t*)take((size_t)Mrows * Hh * 2);

  dim3 gLN(Mrows / 8);
  dim3 gElem(Mrows * Hh / 256);

  // ---- macaron FFN #1: x = hidden; x += 0.5*ffn(LN(x)) ----
  ln_k<<<gLN, 256, 0, stream>>>(hidden, f1_ln_g, f1_ln_b, hln, nullptr, x, Mrows);
  gemm_k<1, false, false><<<dim3(Ff / 128, Mrows / 64), 256, 0, stream>>>(
      hln, f1_w1, f1_b1, t0, nullptr, nullptr, nullptr, nullptr, Mrows, Ff, Hh, 0.f);
  gemm_k<2, false, false><<<dim3(Hh / 128, Mrows / 64), 256, 0, stream>>>(
      t0, f1_w2, f1_b2, nullptr, x, nullptr, nullptr, nullptr, Mrows, Hh, Ff, 0.5f);

  // ---- attention: x += attn(LN(x)) ----
  ln_k<<<gLN, 256, 0, stream>>>(x, at_ln_g, at_ln_b, hln, nullptr, nullptr, Mrows);
  gemm_k<3, false, false><<<dim3(Hh / 128, Mrows / 64), 256, 0, stream>>>(
      hln, wq, bq, qu, nullptr, pos_u, pos_v, qvb, Mrows, Hh, Hh, 0.f);
  gemm_k<0, false, false><<<dim3(Hh / 128, Mrows / 64), 256, 0, stream>>>(
      hln, wk, bk, kbf, nullptr, nullptr, nullptr, nullptr, Mrows, Hh, Hh, 0.f);
  gemm_k<0, false, false><<<dim3(Hh / 128, Mrows / 64), 256, 0, stream>>>(
      hln, wv, bv, vbf, nullptr, nullptr, nullptr, nullptr, Mrows, Hh, Hh, 0.f);
  gemm_k<0, false, true><<<dim3(Hh / 128, (Rpos + 63) / 64), 256, 0, stream>>>(
      posemb, wpos, nullptr, pbf, nullptr, nullptr, nullptr, nullptr, Rpos, Hh, Hh, 0.f);
  attn_k<<<dim3(Ss / 64, Bb * NHh), 128, 0, stream>>>(qu, qvb, kbf, vbf, pbf, ao);
  gemm_k<2, false, false><<<dim3(Hh / 128, Mrows / 64), 256, 0, stream>>>(
      ao, wo, bo, nullptr, x, nullptr, nullptr, nullptr, Mrows, Hh, Hh, 1.f);

  // ---- conv module: x += conv(LN(x)) ----
  ln_k<<<gLN, 256, 0, stream>>>(x, cv_ln_g, cv_ln_b, hln, nullptr, nullptr, Mrows);
  gemm_k<0, true, false><<<dim3((2 * Hh) / 128, Mrows / 64), 256, 0, stream>>>(
      hln, pw1_w, nullptr, t0, nullptr, nullptr, nullptr, nullptr, Mrows, 2 * Hh, Hh, 0.f);
  glu_k<<<gElem, 256, 0, stream>>>(t0, glu);
  dwconv_k<<<gElem, 256, 0, stream>>>(glu, dw_w, bn_g, bn_b, cvo);
  gemm_k<2, true, false><<<dim3(Hh / 128, Mrows / 64), 256, 0, stream>>>(
      cvo, pw2_w, nullptr, nullptr, x, nullptr, nullptr, nullptr, Mrows, Hh, Hh, 1.f);

  // ---- macaron FFN #2 ----
  ln_k<<<gLN, 256, 0, stream>>>(x, f2_ln_g, f2_ln_b, hln, nullptr, nullptr, Mrows);
  gemm_k<1, false, false><<<dim3(Ff / 128, Mrows / 64), 256, 0, stream>>>(
      hln, f2_w1, f2_b1, t0, nullptr, nullptr, nullptr, nullptr, Mrows, Ff, Hh, 0.f);
  gemm_k<2, false, false><<<dim3(Hh / 128, Mrows / 64), 256, 0, stream>>>(
      t0, f2_w2, f2_b2, nullptr, x, nullptr, nullptr, nullptr, Mrows, Hh, Ff, 0.5f);

  // ---- final LN -> d_out (fp32) ----
  ln_k<<<gLN, 256, 0, stream>>>(x, fn_ln_g, fn_ln_b, nullptr, (float*)d_out,
                                nullptr, Mrows);
}
